// InterpreMol_55877524521515
// MI455X (gfx1250) — matrix operand order repaired and checked by reading, not compiled, not run on hardware
//
#include <hip/hip_runtime.h>
#include <hip/hip_bf16.h>
#include <math.h>

// Problem constants
#define BB   64
#define SS   255
#define SP1  256
#define DD   512
#define HH   8
#define LL   6
#define FF_  2048
#define DKK  64
#define HID_ 256
#define MROWS (BB * SP1)   // 16384

typedef __attribute__((ext_vector_type(16))) __bf16 bf16x16;
typedef __attribute__((ext_vector_type(8)))  float  floatx8;

union Frag { unsigned int u[8]; bf16x16 v; };

// ---- CDNA5 async global->LDS path (probe via __has_builtin) -----------------
#if defined(__has_builtin)
#  if __has_builtin(__builtin_amdgcn_global_load_async_to_lds_b128)
#    define HAVE_ASYNC_LDS 1
#  endif
#  if __has_builtin(__builtin_amdgcn_s_wait_asynccnt)
#    define HAVE_WAIT_ASYNC_BUILTIN 1
#  endif
#endif

#ifdef HAVE_ASYNC_LDS
#  ifdef HAVE_WAIT_ASYNC_BUILTIN
#    define WAIT_ASYNC(n) __builtin_amdgcn_s_wait_asynccnt(n)
#  else
#    define WAIT_ASYNC(n) asm volatile("s_wait_asynccnt " #n ::: "memory")
#  endif
#else
#  define WAIT_ASYNC(n) ((void)0)
#endif

// GCC-style int4 vector: matches the builtin's parameter type
// (diagnostic showed: __attribute__((vector_size(4*sizeof(int)))) int AS1 *)
typedef int v4i_gcc __attribute__((vector_size(16)));
typedef __attribute__((address_space(1))) v4i_gcc* g_v4i_ptr;
typedef __attribute__((address_space(3))) v4i_gcc* l_v4i_ptr;

__device__ __forceinline__ void async_copy16(const unsigned short* g, unsigned short* l) {
#ifdef HAVE_ASYNC_LDS
    __builtin_amdgcn_global_load_async_to_lds_b128(
        (g_v4i_ptr)(void*)(unsigned short*)g,
        (l_v4i_ptr)(void*)l, 0, 0);
#else
    *(uint4*)l = *(const uint4*)g;   // sync fallback via VGPRs
#endif
}

__device__ __forceinline__ unsigned short f2bf(float f) {
    unsigned int u = __float_as_uint(f);
    u += 0x7FFFu + ((u >> 16) & 1u);   // round-to-nearest-even
    return (unsigned short)(u >> 16);
}

__device__ __forceinline__ float gelu_exact(float x) {
    return 0.5f * x * (1.0f + erff(x * 0.70710678118654752f));
}

__device__ __forceinline__ floatx8 zero8() {
    floatx8 v;
    #pragma unroll
    for (int i = 0; i < 8; ++i) v[i] = 0.f;
    return v;
}

// ---------------------------------------------------------------------------
// Weight convert + transpose: Wt[n*K + k] = bf16(W[k*N + n])
// ---------------------------------------------------------------------------
__global__ __launch_bounds__(256) void convert_wt_kernel(
    const float* __restrict__ W, unsigned short* __restrict__ Wt, int K, int N)
{
    size_t idx = (size_t)blockIdx.x * 256 + threadIdx.x;
    int k = (int)(idx / N);
    int n = (int)(idx % N);
    Wt[(size_t)n * K + k] = f2bf(W[idx]);
}

// ---------------------------------------------------------------------------
// Embed: x[b,0,:] = cls ; x[b,i,:] = atom_emb[b,i-1,:]; write f32 + bf16
// ---------------------------------------------------------------------------
__global__ __launch_bounds__(256) void embed_kernel(
    const float* __restrict__ atom, const float* __restrict__ cls,
    float* __restrict__ xf, unsigned short* __restrict__ xb)
{
    size_t idx = (size_t)blockIdx.x * 256 + threadIdx.x;
    int d = (int)(idx % DD);
    int m = (int)(idx / DD);
    int b = m >> 8, i = m & 255;
    float v = (i == 0) ? cls[d] : atom[((size_t)b * SS + (i - 1)) * DD + d];
    xf[idx] = v;
    xb[idx] = f2bf(v);
}

// ---------------------------------------------------------------------------
// Generic bf16 WMMA GEMM: C[M,N] = act(A[M,K] @ Wt^T + bias)
//   A : bf16 row-major [M,K];  Bt: bf16 [N,K] (weight, pre-transposed)
//   Outputs: optional f32 Cf, optional bf16 Cb; qkv!=0 scatters Cb to
//   [B,H,SP1,DK] layout.
// Block: 256 threads (8 waves); tile 128x128; wave tile 32x64; K-step 32.
// Double-buffered LDS fed by async global->LDS loads (ASYNCcnt pipeline).
// ---------------------------------------------------------------------------
__global__ __launch_bounds__(256) void gemm_bf16_kernel(
    const unsigned short* __restrict__ A,
    const unsigned short* __restrict__ Bt,
    const float* __restrict__ bias,
    float* __restrict__ Cf,
    unsigned short* __restrict__ Cb,
    int M, int N, int K, int act, int qkv)
{
    __shared__ __align__(16) unsigned short As[2][128][32];
    __shared__ __align__(16) unsigned short Bs[2][128][32];

    int nblk = N >> 7;
    int bx = blockIdx.x % nblk;
    int by = blockIdx.x / nblk;
    int m0 = by << 7, n0 = bx << 7;

    int tid = threadIdx.x;
    int lane = tid & 31, wid = tid >> 5;
    int lrow = lane & 15, halfsel = lane >> 4;
    int wm = (wid & 3) * 32;   // wave row offset in tile
    int wn = (wid >> 2) * 64;  // wave col offset in tile

    // This thread's two 16B chunks per 128x32 tile
    int r0 = tid >> 2,         c0 = (tid & 3) * 8;
    int r1 = (tid + 256) >> 2, c1 = ((tid + 256) & 3) * 8;

    const unsigned short* Ap = A  + (size_t)m0 * K;
    const unsigned short* Bp = Bt + (size_t)n0 * K;

    floatx8 acc[2][4];
    #pragma unroll
    for (int i = 0; i < 2; ++i)
        #pragma unroll
        for (int j = 0; j < 4; ++j) acc[i][j] = zero8();

    // prologue: tile 0 into buffer 0 (4 async ops per thread)
    async_copy16(Ap + (size_t)r0 * K + c0, &As[0][r0][c0]);
    async_copy16(Ap + (size_t)r1 * K + c1, &As[0][r1][c1]);
    async_copy16(Bp + (size_t)r0 * K + c0, &Bs[0][r0][c0]);
    async_copy16(Bp + (size_t)r1 * K + c1, &Bs[0][r1][c1]);

    int nk = K >> 5;
    for (int kt = 0; kt < nk; ++kt) {
        int cur = kt & 1;
        __syncthreads();   // everyone done reading buffer cur^1 (from kt-1)
        if (kt + 1 < nk) {
            int k1 = (kt + 1) << 5;
            async_copy16(Ap + (size_t)r0 * K + k1 + c0, &As[cur ^ 1][r0][c0]);
            async_copy16(Ap + (size_t)r1 * K + k1 + c1, &As[cur ^ 1][r1][c1]);
            async_copy16(Bp + (size_t)r0 * K + k1 + c0, &Bs[cur ^ 1][r0][c0]);
            async_copy16(Bp + (size_t)r1 * K + k1 + c1, &Bs[cur ^ 1][r1][c1]);
            WAIT_ASYNC(4);   // loads retire in order -> tile kt has landed
        } else {
            WAIT_ASYNC(0);
        }
        __syncthreads();     // tile kt visible to all waves

        Frag a[2], b[4];
        int koff = halfsel * 8;
        int kh   = halfsel * 16;
        #pragma unroll
        for (int i = 0; i < 2; ++i) {
            const unsigned int* p = (const unsigned int*)(&As[cur][wm + i * 16 + lrow][0]);
            #pragma unroll
            for (int q = 0; q < 8; ++q) a[i].u[q] = p[(kh + 2 * q) >> 1];
        }
        #pragma unroll
        for (int j = 0; j < 4; ++j) {
            const unsigned int* p = (const unsigned int*)(&Bs[cur][wn + j * 16 + lrow][0]);
            #pragma unroll
            for (int q = 0; q < 8; ++q) b[j].u[q] = p[(kh + 2 * q) >> 1];
        }
        #pragma unroll
        for (int i = 0; i < 2; ++i)
            #pragma unroll
            for (int j = 0; j < 4; ++j)
                acc[i][j] = __builtin_amdgcn_wmma_f32_16x16x32_bf16(
                    false, a[i].v, false, b[j].v, (short)0, acc[i][j], false, false);
    }

    // Epilogue
    #pragma unroll
    for (int j = 0; j < 4; ++j) {
        int n = n0 + wn + j * 16 + lrow;
        float bv = bias ? bias[n] : 0.f;
        #pragma unroll
        for (int i = 0; i < 2; ++i) {
            int mbase = m0 + wm + i * 16 + halfsel * 8;
            #pragma unroll
            for (int r = 0; r < 8; ++r) {
                int m = mbase + r;
                float v = acc[i][j][r] + bv;
                if (act == 1) v = gelu_exact(v);
                if (Cf) Cf[(size_t)m * N + n] = v;
                if (Cb) {
                    size_t oi;
                    if (qkv) {
                        int b8 = m >> 8, ii = m & 255, hh = n >> 6, dd = n & 63;
                        oi = (((size_t)b8 * HH + hh) * SP1 + ii) * DKK + dd;
                    } else {
                        oi = (size_t)m * N + n;
                    }
                    Cb[oi] = f2bf(v);
                }
            }
        }
    }
}

// ---------------------------------------------------------------------------
// Attention (flash-style, online softmax), bf16 WMMA both matmuls.
// Block = 128 threads (4 waves), one (b, h, 64-query tile).
// ---------------------------------------------------------------------------
__global__ __launch_bounds__(128) void attn_kernel(
    const unsigned short* __restrict__ qbf, const unsigned short* __restrict__ kbf,
    const unsigned short* __restrict__ vbf, const float* __restrict__ eb,
    const unsigned char* __restrict__ mask, unsigned short* __restrict__ obf)
{
    __shared__ __align__(16) unsigned short Qs[64][64];      // [i][d]
    __shared__ __align__(16) unsigned short Ks[64][64];      // [j][d]
    __shared__ __align__(16) unsigned short Vt[64][64];      // [d][j]
    __shared__ __align__(16) unsigned short Ps[4][16][64];   // wave-private probs

    int it = blockIdx.x & 3;
    int h  = (blockIdx.x >> 2) & 7;
    int b  = blockIdx.x >> 5;
    int tid = threadIdx.x;
    int lane = tid & 31, wid = tid >> 5;
    int lrow = lane & 15, halfsel = lane >> 4;

    size_t headbase = ((size_t)b * HH + h) * SP1 * DKK;
    const unsigned short* qp = qbf + headbase + (size_t)it * 64 * DKK;
    const unsigned short* kp = kbf + headbase;
    const unsigned short* vp = vbf + headbase;

    // Q tile 64x64 straight copy
    #pragma unroll
    for (int g = 0; g < 4; ++g) {
        int gi = tid + g * 128;
        int rr = gi >> 3, cc = (gi & 7) * 8;
        *(uint4*)(&Qs[rr][cc]) = *(const uint4*)(qp + (size_t)rr * DKK + cc);
    }

    float mrun[8], srun[8];
    floatx8 oacc[4];
    #pragma unroll
    for (int r = 0; r < 8; ++r) { mrun[r] = -__builtin_inff(); srun[r] = 0.f; }
    #pragma unroll
    for (int f = 0; f < 4; ++f) oacc[f] = zero8();

    for (int jt = 0; jt < 4; ++jt) {
        int j0 = jt * 64;
        __syncthreads();
        // K tile straight, V tile transposed
        #pragma unroll
        for (int g = 0; g < 4; ++g) {
            int gi = tid + g * 128;
            int rr = gi >> 3, cc = (gi & 7) * 8;
            *(uint4*)(&Ks[rr][cc]) = *(const uint4*)(kp + (size_t)(j0 + rr) * DKK + cc);
            uint4 vv = *(const uint4*)(vp + (size_t)(j0 + rr) * DKK + cc);
            unsigned short e[8];
            *(uint4*)e = vv;
            #pragma unroll
            for (int q = 0; q < 8; ++q) Vt[cc + q][rr] = e[q];
        }
        __syncthreads();

        // scores: 16 rows x 64 cols per wave
        floatx8 sc[4];
        #pragma unroll
        for (int f = 0; f < 4; ++f) sc[f] = zero8();
        #pragma unroll
        for (int dk0 = 0; dk0 < 64; dk0 += 32) {
            Frag qa;
            {
                const unsigned int* p = (const unsigned int*)(&Qs[wid * 16 + lrow][0]);
                int koff = halfsel * 16;
                #pragma unroll
                for (int q = 0; q < 8; ++q) qa.u[q] = p[(dk0 + koff + 2 * q) >> 1];
            }
            #pragma unroll
            for (int jf = 0; jf < 4; ++jf) {
                Frag kb;
                const unsigned int* p = (const unsigned int*)(&Ks[jf * 16 + lrow][0]);
                int kh = halfsel * 16;
                #pragma unroll
                for (int q = 0; q < 8; ++q) kb.u[q] = p[(dk0 + kh + 2 * q) >> 1];
                sc[jf] = __builtin_amdgcn_wmma_f32_16x16x32_bf16(
                    false, qa.v, false, kb.v, (short)0, sc[jf], false, false);
            }
        }

        // scale + edge bias + padding mask
        #pragma unroll
        for (int jf = 0; jf < 4; ++jf) {
            int j = j0 + jf * 16 + lrow;
            bool mk = (j >= 1) && (mask[(size_t)b * SS + (j - 1)] != 0);
            #pragma unroll
            for (int r = 0; r < 8; ++r) {
                int i = it * 64 + wid * 16 + r + halfsel * 8;
                int ic = i > 0 ? i - 1 : 0;
                int jc = j > 0 ? j - 1 : 0;
                float bv = eb[(((size_t)b * SS + ic) * SS + jc) * HH + h];
                if (i == 0 || j == 0) bv = 0.f;
                float v = sc[jf][r] * 0.125f + bv;
                sc[jf][r] = mk ? -3.402823466e38f : v;
            }
        }

        // online softmax (rows live in 16-lane groups)
        #pragma unroll
        for (int r = 0; r < 8; ++r) {
            float mc = fmaxf(fmaxf(sc[0][r], sc[1][r]), fmaxf(sc[2][r], sc[3][r]));
            #pragma unroll
            for (int off = 1; off < 16; off <<= 1) mc = fmaxf(mc, __shfl_xor(mc, off, 16));
            float mnew = fmaxf(mrun[r], mc);
            float scale = __expf(mrun[r] - mnew);
            mrun[r] = mnew;
            float ps = 0.f;
            #pragma unroll
            for (int jf = 0; jf < 4; ++jf) {
                float p = __expf(sc[jf][r] - mnew);
                sc[jf][r] = p;
                ps += p;
            }
            #pragma unroll
            for (int off = 1; off < 16; off <<= 1) ps += __shfl_xor(ps, off, 16);
            srun[r] = srun[r] * scale + ps;
            #pragma unroll
            for (int df = 0; df < 4; ++df) oacc[df][r] *= scale;
        }

        // probs (bf16) to wave-private LDS
        #pragma unroll
        for (int jf = 0; jf < 4; ++jf)
            #pragma unroll
            for (int r = 0; r < 8; ++r)
                Ps[wid][r + halfsel * 8][jf * 16 + lrow] = f2bf(sc[jf][r]);

        // P @ V
        #pragma unroll
        for (int jk0 = 0; jk0 < 64; jk0 += 32) {
            Frag pa;
            {
                const unsigned int* p = (const unsigned int*)(&Ps[wid][lrow][0]);
                int koff = halfsel * 16;
                #pragma unroll
                for (int q = 0; q < 8; ++q) pa.u[q] = p[(jk0 + koff + 2 * q) >> 1];
            }
            #pragma unroll
            for (int df = 0; df < 4; ++df) {
                Frag vb;
                const unsigned int* p = (const unsigned int*)(&Vt[df * 16 + lrow][0]);
                int kh = halfsel * 16;
                #pragma unroll
                for (int q = 0; q < 8; ++q) vb.u[q] = p[(jk0 + kh + 2 * q) >> 1];
                oacc[df] = __builtin_amdgcn_wmma_f32_16x16x32_bf16(
                    false, pa.v, false, vb.v, (short)0, oacc[df], false, false);
            }
        }
    }

    // normalize and write bf16 to [b, i, h*64+d] (row-major [M, D])
    #pragma unroll
    for (int df = 0; df < 4; ++df) {
        int d = df * 16 + lrow;
        #pragma unroll
        for (int r = 0; r < 8; ++r) {
            int i = it * 64 + wid * 16 + r + halfsel * 8;
            float o = oacc[df][r] / srun[r];
            obf[((size_t)b * SP1 + i) * DD + h * DKK + d] = f2bf(o);
        }
    }
}

// ---------------------------------------------------------------------------
// y = LayerNorm(x + r) * g + b ; write f32 (in-place safe) + bf16
// ---------------------------------------------------------------------------
__global__ __launch_bounds__(256) void add_ln_kernel(
    const float* __restrict__ x, const float* __restrict__ r,
    const float* __restrict__ g, const float* __restrict__ bta,
    float* __restrict__ xo, unsigned short* __restrict__ xb)
{
    int row = blockIdx.x;
    int tid = threadIdx.x;
    int lane = tid & 31, wid = tid >> 5;
    __shared__ float part[8];

    const float* xr = x + (size_t)row * DD;
    const float* rr = r + (size_t)row * DD;
    float v0 = xr[tid] + rr[tid];
    float v1 = xr[tid + 256] + rr[tid + 256];

    float s = v0 + v1;
    #pragma unroll
    for (int off = 1; off < 32; off <<= 1) s += __shfl_xor(s, off, 32);
    if (lane == 0) part[wid] = s;
    __syncthreads();
    float tot = 0.f;
    #pragma unroll
    for (int w = 0; w < 8; ++w) tot += part[w];
    float mean = tot * (1.0f / DD);
    __syncthreads();

    float d0 = v0 - mean, d1 = v1 - mean;
    float q = d0 * d0 + d1 * d1;
    #pragma unroll
    for (int off = 1; off < 32; off <<= 1) q += __shfl_xor(q, off, 32);
    if (lane == 0) part[wid] = q;
    __syncthreads();
    float vtot = 0.f;
    #pragma unroll
    for (int w = 0; w < 8; ++w) vtot += part[w];
    float rstd = rsqrtf(vtot * (1.0f / DD) + 1e-5f);

    float y0 = d0 * rstd * g[tid] + bta[tid];
    float y1 = d1 * rstd * g[tid + 256] + bta[tid + 256];
    float* xro = xo + (size_t)row * DD;
    xro[tid] = y0; xro[tid + 256] = y1;
    unsigned short* xbo = xb + (size_t)row * DD;
    xbo[tid] = f2bf(y0); xbo[tid + 256] = f2bf(y1);
}

// ---------------------------------------------------------------------------
// Head: out[b] = gelu(cls @ W1 + b1) @ W2 + b2   (D=512 -> 256 -> 1)
// ---------------------------------------------------------------------------
__global__ __launch_bounds__(256) void head_kernel(
    const float* __restrict__ xf, const float* __restrict__ W1,
    const float* __restrict__ b1, const float* __restrict__ W2,
    const float* __restrict__ b2, float* __restrict__ out)
{
    int b = blockIdx.x;
    __shared__ float cls[DD];
    __shared__ float part[8];
    int tid = threadIdx.x;
    int lane = tid & 31, wid = tid >> 5;

    const float* xr = xf + (size_t)b * SP1 * DD;   // row b*256 == CLS
    cls[tid] = xr[tid];
    cls[tid + 256] = xr[tid + 256];
    __syncthreads();

    float acc = 0.f;
    for (int d = 0; d < DD; ++d) acc += cls[d] * W1[(size_t)d * HID_ + tid];
    float hv = gelu_exact(acc + b1[tid]);
    float val = hv * W2[tid];
    #pragma unroll
    for (int off = 1; off < 32; off <<= 1) val += __shfl_xor(val, off, 32);
    if (lane == 0) part[wid] = val;
    __syncthreads();
    if (tid == 0) {
        float t = 0.f;
        #pragma unroll
        for (int w = 0; w < 8; ++w) t += part[w];
        out[b] = t + b2[0];
    }
}

// ---------------------------------------------------------------------------
// Host driver
// ---------------------------------------------------------------------------
extern "C" void kernel_launch(void* const* d_in, const int* in_sizes, int n_in,
                              void* d_out, int out_size, void* d_ws, size_t ws_size,
                              hipStream_t stream) {
    (void)in_sizes; (void)n_in; (void)out_size; (void)ws_size;

    const float* atom  = (const float*)d_in[0];
    const float* ebias = (const float*)d_in[1];
    const unsigned char* mask = (const unsigned char*)d_in[2];
    const float* cls   = (const float*)d_in[3];
    const float* Wq = (const float*)d_in[4];   const float* bq = (const float*)d_in[5];
    const float* Wk = (const float*)d_in[6];   const float* bk = (const float*)d_in[7];
    const float* Wv = (const float*)d_in[8];   const float* bv = (const float*)d_in[9];
    const float* Wo = (const float*)d_in[10];  const float* bo = (const float*)d_in[11];
    const float* ln1g = (const float*)d_in[12]; const float* ln1b = (const float*)d_in[13];
    const float* W1 = (const float*)d_in[14];  const float* b1 = (const float*)d_in[15];
    const float* W2 = (const float*)d_in[16];  const float* b2 = (const float*)d_in[17];
    const float* ln2g = (const float*)d_in[18]; const float* ln2b = (const float*)d_in[19];
    const float* hW1 = (const float*)d_in[20]; const float* hb1 = (const float*)d_in[21];
    const float* hW2 = (const float*)d_in[22]; const float* hb2 = (const float*)d_in[23];
    float* out = (float*)d_out;

    unsigned char* base = (unsigned char*)d_ws;
    size_t off = 0;
    auto alloc = [&](size_t bytes) -> void* {
        void* p = base + off;
        off += (bytes + 255) & ~(size_t)255;
        return p;
    };
    float*          xf  = (float*)alloc((size_t)MROWS * DD * 4);
    unsigned short* xb  = (unsigned short*)alloc((size_t)MROWS * DD * 2);
    unsigned short* qb  = (unsigned short*)alloc((size_t)MROWS * DD * 2);
    unsigned short* kb  = (unsigned short*)alloc((size_t)MROWS * DD * 2);
    unsigned short* vb  = (unsigned short*)alloc((size_t)MROWS * DD * 2);
    unsigned short* ab  = (unsigned short*)alloc((size_t)MROWS * DD * 2);
    float*          pf  = (float*)alloc((size_t)MROWS * DD * 4);
    unsigned short* hb  = (unsigned short*)alloc((size_t)MROWS * FF_ * 2);
    unsigned short* wqT = (unsigned short*)alloc((size_t)LL * DD * DD * 2);
    unsigned short* wkT = (unsigned short*)alloc((size_t)LL * DD * DD * 2);
    unsigned short* wvT = (unsigned short*)alloc((size_t)LL * DD * DD * 2);
    unsigned short* woT = (unsigned short*)alloc((size_t)LL * DD * DD * 2);
    unsigned short* w1T = (unsigned short*)alloc((size_t)LL * DD * FF_ * 2);
    unsigned short* w2T = (unsigned short*)alloc((size_t)LL * FF_ * DD * 2);

    // 1. convert all weights to bf16 transposed [N,K]
    for (int l = 0; l < LL; ++l) {
        convert_wt_kernel<<<(DD * DD) / 256, 256, 0, stream>>>(Wq + (size_t)l * DD * DD, wqT + (size_t)l * DD * DD, DD, DD);
        convert_wt_kernel<<<(DD * DD) / 256, 256, 0, stream>>>(Wk + (size_t)l * DD * DD, wkT + (size_t)l * DD * DD, DD, DD);
        convert_wt_kernel<<<(DD * DD) / 256, 256, 0, stream>>>(Wv + (size_t)l * DD * DD, wvT + (size_t)l * DD * DD, DD, DD);
        convert_wt_kernel<<<(DD * DD) / 256, 256, 0, stream>>>(Wo + (size_t)l * DD * DD, woT + (size_t)l * DD * DD, DD, DD);
        convert_wt_kernel<<<(DD * FF_) / 256, 256, 0, stream>>>(W1 + (size_t)l * DD * FF_, w1T + (size_t)l * DD * FF_, DD, FF_);
        convert_wt_kernel<<<(FF_ * DD) / 256, 256, 0, stream>>>(W2 + (size_t)l * FF_ * DD, w2T + (size_t)l * FF_ * DD, FF_, DD);
    }

    // 2. embed (CLS + atoms) -> xf, xb
    embed_kernel<<<(MROWS * DD) / 256, 256, 0, stream>>>(atom, cls, xf, xb);

    int gProj = (MROWS / 128) * (DD / 128);   // 512 blocks
    int gFF1  = (MROWS / 128) * (FF_ / 128);  // 2048 blocks

    for (int l = 0; l < LL; ++l) {
        // QKV projections (bf16 output, head-split layout)
        gemm_bf16_kernel<<<gProj, 256, 0, stream>>>(xb, wqT + (size_t)l * DD * DD, bq + (size_t)l * DD,
                                                    nullptr, qb, MROWS, DD, DD, 0, 1);
        gemm_bf16_kernel<<<gProj, 256, 0, stream>>>(xb, wkT + (size_t)l * DD * DD, bk + (size_t)l * DD,
                                                    nullptr, kb, MROWS, DD, DD, 0, 1);
        gemm_bf16_kernel<<<gProj, 256, 0, stream>>>(xb, wvT + (size_t)l * DD * DD, bv + (size_t)l * DD,
                                                    nullptr, vb, MROWS, DD, DD, 0, 1);
        // attention
        attn_kernel<<<BB * HH * (SP1 / 64), 128, 0, stream>>>(qb, kb, vb, ebias, mask, ab);
        // output projection
        gemm_bf16_kernel<<<gProj, 256, 0, stream>>>(ab, woT + (size_t)l * DD * DD, bo + (size_t)l * DD,
                                                    pf, nullptr, MROWS, DD, DD, 0, 0);
        // x = LN(x + attn_out)
        add_ln_kernel<<<MROWS, 256, 0, stream>>>(xf, pf, ln1g + (size_t)l * DD, ln1b + (size_t)l * DD, xf, xb);
        // FFN
        gemm_bf16_kernel<<<gFF1, 256, 0, stream>>>(xb, w1T + (size_t)l * DD * FF_, b1 + (size_t)l * FF_,
                                                   nullptr, hb, MROWS, FF_, DD, 1, 0);
        gemm_bf16_kernel<<<gProj, 256, 0, stream>>>(hb, w2T + (size_t)l * FF_ * DD, b2 + (size_t)l * DD,
                                                    pf, nullptr, MROWS, DD, FF_, 0, 0);
        // x = LN(x + ff)
        add_ln_kernel<<<MROWS, 256, 0, stream>>>(xf, pf, ln2g + (size_t)l * DD, ln2b + (size_t)l * DD, xf, xb);
    }

    // 3. head on CLS token
    head_kernel<<<BB, 256, 0, stream>>>(xf, hW1, hb1, hW2, hb2, out);
}
